// CUFLayer_83640193122985
// MI455X (gfx1250) — matrix operand order repaired and checked by reading, not compiled, not run on hardware
//
#include <hip/hip_runtime.h>
#include <hip/hip_bf16.h>

typedef __attribute__((ext_vector_type(16))) __bf16 v16bf;
typedef __attribute__((ext_vector_type(2)))  __bf16 v2bf;
typedef __attribute__((ext_vector_type(8)))  float  v8f;

#define C_IN   256
#define F_OUT  128
#define HT     128
#define WT     128
#define HIN    64
#define WIN    64
#define NU     32
#define NOUT   2304          // 9 * 256
#define NBLK   144           // 2304 / 16
#define SSTR   264           // padded row stride (ushorts) for s tile in LDS

__device__ __forceinline__ unsigned short f2bf(float x) {
  unsigned int u = __builtin_bit_cast(unsigned int, x);
  u += 0x7fffu + ((u >> 16) & 1u);              // round-to-nearest-even
  return (unsigned short)(u >> 16);
}
__device__ __forceinline__ float bf2f(unsigned short h) {
  unsigned int u = ((unsigned int)h) << 16;
  return __builtin_bit_cast(float, u);
}
// packed f32x2 -> bf16x2 (v_cvt_pk_bf16_f32 when available; RNE either way)
__device__ __forceinline__ unsigned int pack2(float lo, float hi) {
#if __has_builtin(__builtin_amdgcn_cvt_pk_bf16_f32)
  v2bf r = __builtin_amdgcn_cvt_pk_bf16_f32(lo, hi);
  return __builtin_bit_cast(unsigned int, r);
#else
  return (unsigned int)f2bf(lo) | ((unsigned int)f2bf(hi) << 16);
#endif
}

union Frag16 { unsigned int u[8]; v16bf v; };

// ---------------------------------------------------------------------------
// Kernel 1: hyper-network trunk. One thread per pixel of the 128x128 grid.
// 118 DCT features -> 4 x dense(32)+ReLU. Writes bf16 hidden state [16384,32].
// ~0.1 GFLOP total: scalar fp32 VALU is fine; keeps full precision in trunk.
// ---------------------------------------------------------------------------
__global__ __launch_bounds__(256) void hyper_mlp(
    const float* __restrict__ W1, const float* __restrict__ b1,
    const float* __restrict__ W2, const float* __restrict__ b2,
    const float* __restrict__ W3, const float* __restrict__ b3,
    const float* __restrict__ W4, const float* __restrict__ b4,
    unsigned short* __restrict__ h_out)
{
  const int pix = blockIdx.x * blockDim.x + threadIdx.x;   // 0..16383
  const int hr = pix >> 7, wc = pix & 127;
  const float PI = 3.14159265358979323846f;

  float acc[NU];
#pragma unroll
  for (int u = 0; u < NU; ++u) acc[u] = b1[u];

  int idx = 0;
  auto addf = [&](float feat) {
    const float* wr = W1 + idx * NU;
#pragma unroll
    for (int u = 0; u < NU; ++u) acc[u] += feat * wr[u];
    ++idx;
  };

  // delta_enc: cos(pi*(2x+1)*f), f = linspace(1,2,25), x in {h/127, w/127}
  {
    float xs[2] = { hr * (1.0f / 127.0f), wc * (1.0f / 127.0f) };
    for (int d = 0; d < 2; ++d) {
      float base = PI * (2.0f * xs[d] + 1.0f);
      for (int j = 0; j < 25; ++j)
        addf(cosf(base * (1.0f + (float)j * (1.0f / 24.0f))));
    }
  }
  // scale_enc: x = 2.0 for both components (H_t/H_in = W_t/W_in = 2)
  {
    float base = PI * 5.0f;
    for (int d = 0; d < 2; ++d)
      for (int j = 0; j < 25; ++j)
        addf(cosf(base * (1.0f + (float)j * (1.0f / 24.0f))));
  }
  // kernel_enc: mean over the 9 taps of cos(pi*(2k+1)), f = linspace(1,1,9)
  for (int d = 0; d < 2; ++d)
    for (int j = 0; j < 9; ++j) {
      float s = 0.f;
      for (int p = 0; p < 9; ++p) {
        float kv = (float)((d == 0) ? (p / 3) : (p % 3)) - 1.0f;
        s += cosf(PI * (2.0f * kv + 1.0f));
      }
      addf(s * (1.0f / 9.0f));
    }

  // hidden layers 2..4
  const float* Ws[3] = { W2, W3, W4 };
  const float* bs[3] = { b2, b3, b4 };
  float tmp[NU];
  for (int L = 0; L < 3; ++L) {
#pragma unroll
    for (int u = 0; u < NU; ++u) acc[u] = fmaxf(acc[u], 0.0f);
    const float* Wl = Ws[L]; const float* bl = bs[L];
    for (int v = 0; v < NU; ++v) {
      float s = bl[v];
#pragma unroll
      for (int u = 0; u < NU; ++u) s += acc[u] * Wl[u * NU + v];
      tmp[v] = s;
    }
#pragma unroll
    for (int v = 0; v < NU; ++v) acc[v] = tmp[v];
  }
  // store bf16, packed conversion two-at-a-time
  unsigned int* hp = (unsigned int*)(h_out + (size_t)pix * NU);
#pragma unroll
  for (int u = 0; u < NU; u += 2)
    hp[u >> 1] = pack2(fmaxf(acc[u], 0.0f), fmaxf(acc[u + 1], 0.0f));
}

// ---------------------------------------------------------------------------
// Weight pre-swizzle into the B-fragment order used here (16-bit B 32x16, contiguous K as in A):
//   lane l: N = l&15, K-half = (l<16 ? 0 : 16); VGPR v: packed K = 2v, 2v+1.
// Each consuming lane then loads its 8 fragment dwords contiguously (b128).
// ---------------------------------------------------------------------------
__global__ void prep_wout(const float* __restrict__ W_out,
                          unsigned int* __restrict__ swz)
{
  int t = blockIdx.x * blockDim.x + threadIdx.x;      // NBLK*256
  if (t >= NBLK * 256) return;
  int v = t & 7, lane = (t >> 3) & 31, blk = t >> 8;
  int n  = blk * 16 + (lane & 15);
  int k0 = 2 * v + ((lane < 16) ? 0 : 16);
  swz[t] = pack2(W_out[k0 * NOUT + n], W_out[(k0 + 1) * NOUT + n]);
}

__global__ void prep_wproj(const float* __restrict__ W_proj,
                           unsigned int* __restrict__ swz)
{
  int t = blockIdx.x * blockDim.x + threadIdx.x;      // 8 fb * 8 kc * 256
  if (t >= 64 * 256) return;
  int v = t & 7, lane = (t >> 3) & 31, kc = (t >> 8) & 7, fb = t >> 11;
  int f  = fb * 16 + (lane & 15);
  int c0 = kc * 32 + 2 * v + ((lane < 16) ? 0 : 16);
  swz[t] = pack2(W_proj[c0 * F_OUT + f], W_proj[(c0 + 1) * F_OUT + f]);
}

// ---------------------------------------------------------------------------
// Fused kernel: one workgroup per (row, 16-col) tile, 256 threads = 8 waves.
//  A) 144 x v_wmma_f32_16x16x32_bf16 build the 16x2304 bf16 kernel slab in
//     LDS (18 WMMAs / wave), +b_out. Computed ONCE, reused for all 4 batches.
//  B) per batch: 9-tap weighted patch sum with the NN upsample folded into
//     index math (src = dst>>1); unique footprint (~30 KB/tile) is served by
//     WGP$, so plain cached loads beat any dedup VALU. Result tile s[16][256]
//     goes to LDS as bf16 (row-padded against bank conflicts).
//  C) per batch: projection GEMM [16x256]x[256x128]; each wave owns one
//     16-wide F block, 8 accumulating WMMAs over K, +b_proj, coalesced store.
// LDS: 16*2304*2 + 16*SSTR*2 = 82,176 B (WGP has 320 KB).
// ---------------------------------------------------------------------------
__global__ __launch_bounds__(256) void fused_conv(
    const float* __restrict__ xin,                 // [4,64,64,256]
    const unsigned short* __restrict__ h_ws,       // [16384,32] bf16
    const unsigned int* __restrict__ wout_swz,
    const float* __restrict__ b_out,
    const unsigned int* __restrict__ wproj_swz,
    const float* __restrict__ b_proj,
    float* __restrict__ out)                       // [4,128,128,128]
{
  extern __shared__ unsigned short lds[];
  unsigned short* KernL = lds;                     // [16][2304]
  unsigned short* sL    = lds + 16 * NOUT;         // [16][SSTR]

  const int tid   = threadIdx.x;
  const int lane  = tid & 31;
  const int wav   = tid >> 5;
  const int hrow  = blockIdx.y;
  const int w0    = blockIdx.x * 16;
  const int pixbase = hrow * WT + w0;
  const int M     = lane & 15;
  const int hhalf = (lane < 16) ? 0 : 16;          // A-frag K-half offset

  // ---- A fragment: hidden state for this tile's 16 pixels (16x32 bf16).
  // A layout used here (contiguous K): lane<16 -> K {0..15}; lane>=16 -> K {16..31};
  // K pairs are even-based, so each dword comes from one aligned u32 load.
  Frag16 afr;
#pragma unroll
  for (int v = 0; v < 8; ++v) {
    int k0 = 2 * v + hhalf;
    afr.u[v] = *(const unsigned int*)(h_ws + (pixbase + M) * NU + k0);
  }

  // ---- Step A: kernel slab GEMM, 18 N-blocks per wave
#pragma unroll 1
  for (int i = 0; i < 18; ++i) {
    int blk = wav * 18 + i;
    int n0  = blk * 16;
    Frag16 bfr;
    const unsigned int* bp = wout_swz + blk * 256 + lane * 8;
#pragma unroll
    for (int v = 0; v < 8; ++v) bfr.u[v] = bp[v];
    v8f d = {};
    d = __builtin_amdgcn_wmma_f32_16x16x32_bf16(
            false, afr.v, false, bfr.v, (short)0, d, false, false);
    float bo = b_out[n0 + (lane & 15)];
    const int n = n0 + (lane & 15);
    // rows v, v+1 converted as a pair (v_cvt_pk_bf16_f32), split stores:
    // low half -> ds_store_b16, high half -> ds_store_b16_d16_hi
#pragma unroll
    for (int v = 0; v < 8; v += 2) {
      unsigned int pk = pack2(d[v] + bo, d[v + 1] + bo);
      int r0 = (lane < 16) ? v : (v + 8);          // C/D layout: M = v (+8)
      KernL[r0 * NOUT + n]       = (unsigned short)(pk & 0xffffu);
      KernL[(r0 + 1) * NOUT + n] = (unsigned short)(pk >> 16);
    }
  }
  __syncthreads();

  const int p     = tid >> 4;                      // pixel in tile
  const int c0    = (tid & 15) * 16;               // channel chunk base
  const int wglob = w0 + p;

  for (int b = 0; b < 4; ++b) {
    // ---- Step B: 3x3 SAME weighted patch sum (NN upsample = index >>1)
    float s[16];
#pragma unroll
    for (int cc = 0; cc < 16; ++cc) s[cc] = 0.f;
#pragma unroll
    for (int di = 0; di < 3; ++di) {
      int hy = hrow + di - 1;
      if ((unsigned)hy >= (unsigned)HT) continue;  // zero pad
      int sy = hy >> 1;
#pragma unroll
      for (int dj = 0; dj < 3; ++dj) {
        int wx = wglob + dj - 1;
        if ((unsigned)wx >= (unsigned)WT) continue;
        int sx = wx >> 1;
        const float* mp =
            xin + (((size_t)(b * HIN + sy) * WIN + sx) * C_IN + c0);
        // prefetch the same source row of the NEXT batch into cache
        if (b < 3 && di == 1 && dj == 1)
          __builtin_prefetch(mp + (size_t)HIN * WIN * C_IN, 0, 0);
        const unsigned int* kp = (const unsigned int*)
            (KernL + p * NOUT + (di * 3 + dj) * C_IN + c0);
#pragma unroll
        for (int q = 0; q < 8; ++q) {
          unsigned int kk = kp[q];
          s[2 * q]     += mp[2 * q]     * bf2f((unsigned short)(kk & 0xffffu));
          s[2 * q + 1] += mp[2 * q + 1] * bf2f((unsigned short)(kk >> 16));
        }
      }
    }
    unsigned int* sdst = (unsigned int*)(sL + p * SSTR + c0);
#pragma unroll
    for (int q = 0; q < 8; ++q) sdst[q] = pack2(s[2 * q], s[2 * q + 1]);
    __syncthreads();

    // ---- Step C: projection GEMM, wave 'wav' owns F block [wav*16, +16)
    v8f acc = {};
#pragma unroll 1
    for (int kc = 0; kc < 8; ++kc) {
      Frag16 av, bv;
#pragma unroll
      for (int v = 0; v < 8; ++v) {
        int k0 = kc * 32 + 2 * v + hhalf;
        av.u[v] = *(const unsigned int*)(sL + M * SSTR + k0);
      }
      const unsigned int* bp = wproj_swz + ((wav * 8 + kc) * 32 + lane) * 8;
#pragma unroll
      for (int v = 0; v < 8; ++v) bv.u[v] = bp[v];
      acc = __builtin_amdgcn_wmma_f32_16x16x32_bf16(
                false, av.v, false, bv.v, (short)0, acc, false, false);
    }
    float bpv = b_proj[wav * 16 + (lane & 15)];
#pragma unroll
    for (int v = 0; v < 8; ++v) {
      int row = (lane < 16) ? v : (v + 8);
      out[(((size_t)b * HT + hrow) * WT + (w0 + row)) * F_OUT
          + wav * 16 + (lane & 15)] = acc[v] + bpv;
    }
    __syncthreads();                               // sL reused next batch
  }
}

// ---------------------------------------------------------------------------
extern "C" void kernel_launch(void* const* d_in, const int* in_sizes, int n_in,
                              void* d_out, int out_size, void* d_ws, size_t ws_size,
                              hipStream_t stream)
{
  (void)in_sizes; (void)n_in; (void)out_size; (void)ws_size;
  const float* xin    = (const float*)d_in[0];
  // d_in[1] = skip_connection: only its shape matters in the reference.
  const float* W1     = (const float*)d_in[2];
  const float* b1     = (const float*)d_in[3];
  const float* W2     = (const float*)d_in[4];
  const float* b2     = (const float*)d_in[5];
  const float* W3     = (const float*)d_in[6];
  const float* b3     = (const float*)d_in[7];
  const float* W4     = (const float*)d_in[8];
  const float* b4     = (const float*)d_in[9];
  const float* W_out  = (const float*)d_in[10];
  const float* b_out  = (const float*)d_in[11];
  const float* W_proj = (const float*)d_in[12];
  const float* b_proj = (const float*)d_in[13];
  float* out = (float*)d_out;

  // workspace: h (1 MiB bf16) | W_out swizzled (144 KiB) | W_proj swizzled (64 KiB)
  char* ws = (char*)d_ws;
  unsigned short* h_ws     = (unsigned short*)ws;
  unsigned int*   wout_swz = (unsigned int*)(ws + (size_t)16384 * NU * 2);
  unsigned int*   wproj_swz =
      (unsigned int*)(ws + (size_t)16384 * NU * 2 + (size_t)NBLK * 256 * 4);

  hyper_mlp<<<64, 256, 0, stream>>>(W1, b1, W2, b2, W3, b3, W4, b4, h_ws);
  prep_wout<<<NBLK, 256, 0, stream>>>(W_out, wout_swz);
  prep_wproj<<<64, 256, 0, stream>>>(W_proj, wproj_swz);

  size_t shmem = (size_t)(16 * NOUT + 16 * SSTR) * sizeof(unsigned short);
  dim3 grid(WT / 16, HT);                          // (8, 128) tiles
  fused_conv<<<grid, 256, shmem, stream>>>(
      xin, h_ws, wout_swz, b_out, wproj_swz, b_proj, out);
}
